// MambaModel_87540023427862
// MI455X (gfx1250) — matrix operand order repaired and checked by reading, not compiled, not run on hardware
//
#include <hip/hip_runtime.h>

// ---------------------------------------------------------------------------
// Mamba forward for MI455X / gfx1250 (wave32, WMMA bf16 + async-LDS path).
// ---------------------------------------------------------------------------

typedef __bf16 bf16;
typedef __attribute__((ext_vector_type(16))) __bf16 bf16x16;
typedef __attribute__((ext_vector_type(8)))  float  f32x8;

static constexpr int VOCAB   = 32000;
static constexpr int D_MODEL = 1024;
static constexpr int D_STATE = 16;
static constexpr int D_INNER = 2048;
static constexpr int DT_RANK = 64;
static constexpr int BATCH   = 2;
static constexpr int SEQ     = 2048;
static constexpr int M_TOK   = BATCH * SEQ;          // 4096 token rows

// ---------------------------------------------------------------------------
// small helpers
// ---------------------------------------------------------------------------
__device__ __forceinline__ float siluf(float x) {
    return x * (1.0f / (1.0f + __expf(-x)));
}
__device__ __forceinline__ float softplusf(float x) {
    return (x > 20.0f) ? x : log1pf(__expf(x));
}

// ---------------------------------------------------------------------------
// f32 -> bf16 conversion (weights & activations)
// ---------------------------------------------------------------------------
__global__ __launch_bounds__(256) void cvt_bf16_kernel(const float* __restrict__ src,
                                                       bf16* __restrict__ dst, int n) {
    int i = blockIdx.x * 256 + threadIdx.x;
    if (i < n) dst[i] = (bf16)src[i];
}

// ---------------------------------------------------------------------------
// embedding gather -> bf16 activations
// ---------------------------------------------------------------------------
__global__ __launch_bounds__(256) void embed_kernel(const int* __restrict__ tokens,
                                                    const float* __restrict__ emb,
                                                    bf16* __restrict__ u) {
    int idx = blockIdx.x * 256 + threadIdx.x;          // over M_TOK * D_MODEL
    if (idx >= M_TOK * D_MODEL) return;
    int k = idx & (D_MODEL - 1);
    int m = idx >> 10;
    int tok = tokens[m];
    u[idx] = (bf16)emb[(size_t)tok * D_MODEL + k];
}

// ---------------------------------------------------------------------------
// Tiled WMMA GEMM:  C[M,N] = A[M,K] * W[N,K]^T  (+bias, +softplus)
//   A, W bf16 row-major; C f32 row-major.
//   Block tile 128x128, BK=32; 8 waves, wave tile 64x32 (4x2 WMMA frags).
//   ASYNCPATH=true : N % 128 == 0, tiles staged via global_load_async_to_lds
//                    with LDS double buffering (ASYNCcnt-tracked).
//   ASYNCPATH=false: guarded synchronous staging (x_proj, N=96).
// ---------------------------------------------------------------------------
static constexpr int BM = 128, BN = 128, BK = 32;
static constexpr int LDSK = BK + 8;                    // pad: 80B row stride

template <int EPI, bool ASYNCPATH>  // EPI 0: none, 1: +bias, 2: +bias,softplus
__global__ __launch_bounds__(256)
void wmma_gemm_kernel(const bf16* __restrict__ A, int lda,
                      const bf16* __restrict__ W, int ldb,
                      const float* __restrict__ bias,
                      float* __restrict__ C, int ldc,
                      int M, int N, int K) {
    __shared__ bf16 As[2][BM][LDSK];
    __shared__ bf16 Bs[2][BN][LDSK];

    const int tid    = threadIdx.x;
    const int wid    = tid >> 5;
    const int lane   = tid & 31;
    const int half   = lane >> 4;        // 0: lanes 0-15, 1: lanes 16-31
    const int r      = lane & 15;
    const int wave_m = (wid >> 2) * 64;  // 0 / 64
    const int wave_n = (wid & 3) * 32;   // 0 / 32 / 64 / 96

    const int m0 = blockIdx.y * BM;
    const int n0 = blockIdx.x * BN;

    f32x8 acc[4][2];
#pragma unroll
    for (int i = 0; i < 4; ++i)
#pragma unroll
        for (int j = 0; j < 2; ++j) acc[i][j] = (f32x8){0.f,0.f,0.f,0.f,0.f,0.f,0.f,0.f};

    // cooperative tile loaders: 256 threads, 2 threads per row, 16 bf16 each
    const int lrow = tid >> 1;           // 0..127
    const int lcol = (tid & 1) * 16;     // 0 / 16
    const int nrow = n0 + lrow;

    const bf16* aptr = A + (size_t)(m0 + lrow) * lda + lcol;
    const bf16* bptr = W + (size_t)nrow * ldb + lcol;

    union FragU { bf16x16 v; uint4 q[2]; };

    if constexpr (ASYNCPATH) {
        // LDS byte offsets for this thread's slice of each buffer
        unsigned lds_a0 = (unsigned)(uintptr_t)&As[0][lrow][lcol];
        unsigned lds_a1 = (unsigned)(uintptr_t)&As[1][lrow][lcol];
        unsigned lds_b0 = (unsigned)(uintptr_t)&Bs[0][lrow][lcol];
        unsigned lds_b1 = (unsigned)(uintptr_t)&Bs[1][lrow][lcol];

        auto stage = [&](int buf, int k0) {
            unsigned la = buf ? lds_a1 : lds_a0;
            unsigned lb = buf ? lds_b1 : lds_b0;
            unsigned long long ga = (unsigned long long)(uintptr_t)(aptr + k0);
            unsigned long long gb = (unsigned long long)(uintptr_t)(bptr + k0);
            // each lane copies 2x16B straight into LDS; offset applies to both
            // global and LDS address (ISA 08_async_tensor 4.4)
            asm volatile(
                "global_load_async_to_lds_b128 %0, %2, off\n\t"
                "global_load_async_to_lds_b128 %0, %2, off offset:16\n\t"
                "global_load_async_to_lds_b128 %1, %3, off\n\t"
                "global_load_async_to_lds_b128 %1, %3, off offset:16"
                :: "v"(la), "v"(lb), "v"(ga), "v"(gb)
                : "memory");
        };

        stage(0, 0);
        asm volatile("s_wait_asynccnt 0x0" ::: "memory");
        __syncthreads();

        for (int k0 = 0; k0 < K; k0 += BK) {
            const int cur = (k0 / BK) & 1;
            if (k0 + BK < K) stage(cur ^ 1, k0 + BK);   // overlap next tile DMA

            const bf16 (*Ac)[LDSK] = As[cur];
            const bf16 (*Bc)[LDSK] = Bs[cur];

            FragU fb[2];
#pragma unroll
            for (int j = 0; j < 2; ++j) {
                const bf16* p = &Bc[wave_n + j * 16 + r][half * 16];
                fb[j].q[0] = *(const uint4*)(p);
                fb[j].q[1] = *(const uint4*)(p + 8);
            }
#pragma unroll
            for (int i = 0; i < 4; ++i) {
                FragU fa;
                const bf16* pa = &Ac[wave_m + i * 16 + r][half * 16];
                fa.q[0] = *(const uint4*)(pa);
                fa.q[1] = *(const uint4*)(pa + 8);
#pragma unroll
                for (int j = 0; j < 2; ++j) {
                    acc[i][j] = __builtin_amdgcn_wmma_f32_16x16x32_bf16(
                        false, fa.v, false, fb[j].v, (short)0, acc[i][j],
                        false, false);
                }
            }
            // own async copies into nxt done; barrier makes all slices visible
            // and licenses overwriting cur next iteration
            asm volatile("s_wait_asynccnt 0x0" ::: "memory");
            __syncthreads();
        }
    } else {
        // guarded synchronous staging (only used for N=96 x_proj)
        for (int k0 = 0; k0 < K; k0 += BK) {
            uint4 a0 = *(const uint4*)(aptr + k0);
            uint4 a1 = *(const uint4*)(aptr + k0 + 8);
            *(uint4*)&As[0][lrow][lcol]     = a0;
            *(uint4*)&As[0][lrow][lcol + 8] = a1;

            uint4 b0 = {0u,0u,0u,0u}, b1 = {0u,0u,0u,0u};
            if (nrow < N) {
                b0 = *(const uint4*)(bptr + k0);
                b1 = *(const uint4*)(bptr + k0 + 8);
            }
            *(uint4*)&Bs[0][lrow][lcol]     = b0;
            *(uint4*)&Bs[0][lrow][lcol + 8] = b1;

            if (k0 + BK < K) {
                __builtin_prefetch(aptr + k0 + BK, 0, 3);
                if (nrow < N) __builtin_prefetch(bptr + k0 + BK, 0, 3);
            }
            __syncthreads();

            FragU fb[2];
#pragma unroll
            for (int j = 0; j < 2; ++j) {
                const bf16* p = &Bs[0][wave_n + j * 16 + r][half * 16];
                fb[j].q[0] = *(const uint4*)(p);
                fb[j].q[1] = *(const uint4*)(p + 8);
            }
#pragma unroll
            for (int i = 0; i < 4; ++i) {
                FragU fa;
                const bf16* pa = &As[0][wave_m + i * 16 + r][half * 16];
                fa.q[0] = *(const uint4*)(pa);
                fa.q[1] = *(const uint4*)(pa + 8);
#pragma unroll
                for (int j = 0; j < 2; ++j) {
                    acc[i][j] = __builtin_amdgcn_wmma_f32_16x16x32_bf16(
                        false, fa.v, false, fb[j].v, (short)0, acc[i][j],
                        false, false);
                }
            }
            __syncthreads();
        }
    }

    // ---- epilogue: C/D layout -> row = base + v + half*8, col = r
#pragma unroll
    for (int i = 0; i < 4; ++i) {
#pragma unroll
        for (int j = 0; j < 2; ++j) {
            int row = m0 + wave_m + i * 16 + half * 8;
            int col = n0 + wave_n + j * 16 + r;
            if (col < N) {
                float bv = (EPI >= 1) ? bias[col] : 0.0f;
#pragma unroll
                for (int v = 0; v < 8; ++v) {
                    float val = acc[i][j][v] + bv;
                    if (EPI == 2) val = softplusf(val);
                    C[(size_t)(row + v) * ldc + col] = val;
                }
            }
        }
    }
}

// ---------------------------------------------------------------------------
// causal depthwise conv (K=4) + SiLU over the x half of xz
// ---------------------------------------------------------------------------
__global__ __launch_bounds__(256) void conv_silu_kernel(const float* __restrict__ xz,
                                                        const float* __restrict__ conv_w,
                                                        const float* __restrict__ conv_b,
                                                        float* __restrict__ xconv,
                                                        bf16* __restrict__ x_bf16) {
    int idx = blockIdx.x * 256 + threadIdx.x;          // over M_TOK * D_INNER
    if (idx >= M_TOK * D_INNER) return;
    int d = idx & (D_INNER - 1);
    int m = idx >> 11;                                  // token row
    int l = m & (SEQ - 1);                              // pos within sequence
    float acc = conv_b[d];
#pragma unroll
    for (int k = 0; k < 4; ++k) {
        int li = l - 3 + k;
        if (li >= 0)
            acc += xz[(size_t)(m - 3 + k) * (2 * D_INNER) + d] * conv_w[d * 4 + k];
    }
    float s = siluf(acc);
    xconv[idx]  = s;
    x_bf16[idx] = (bf16)s;
}

// ---------------------------------------------------------------------------
// selective scan: 16 lanes per channel (one per state), sequential over L.
// ---------------------------------------------------------------------------
__global__ __launch_bounds__(256) void scan_kernel(const float* __restrict__ dt,
                                                   const float* __restrict__ xconv,
                                                   const float* __restrict__ xdbl,
                                                   const float* __restrict__ A_log,
                                                   float* __restrict__ y) {
    int s  = threadIdx.x & 15;                          // state index
    int ch = blockIdx.x * 16 + (threadIdx.x >> 4);      // 0 .. BATCH*D_INNER-1
    int b  = ch >> 11;
    int d  = ch & (D_INNER - 1);
    float Aval = -__expf(A_log[d * D_STATE + s]);
    float h = 0.0f;
    const size_t rowbase = (size_t)b * SEQ;
    for (int t = 0; t < SEQ; ++t) {
        size_t row = rowbase + t;
        float dtv = dt[row * D_INNER + d];
        float xv  = xconv[row * D_INNER + d];
        float Bv  = xdbl[row * 96 + DT_RANK + s];
        float Cv  = xdbl[row * 96 + DT_RANK + D_STATE + s];
        h = __expf(dtv * Aval) * h + dtv * Bv * xv;
        float p = h * Cv;
        p += __shfl_xor(p, 1, 32);
        p += __shfl_xor(p, 2, 32);
        p += __shfl_xor(p, 4, 32);
        p += __shfl_xor(p, 8, 32);
        if (s == 0) y[row * D_INNER + d] = p;
    }
}

// ---------------------------------------------------------------------------
// gate: y = (y_scan + D * x_conv) * silu(z)  -> bf16 for out_proj
// ---------------------------------------------------------------------------
__global__ __launch_bounds__(256) void gate_kernel(const float* __restrict__ y_scan,
                                                   const float* __restrict__ xconv,
                                                   const float* __restrict__ xz,
                                                   const float* __restrict__ Dvec,
                                                   bf16* __restrict__ y_bf16) {
    int idx = blockIdx.x * 256 + threadIdx.x;           // over M_TOK * D_INNER
    if (idx >= M_TOK * D_INNER) return;
    int d = idx & (D_INNER - 1);
    int m = idx >> 11;
    float z = xz[(size_t)m * (2 * D_INNER) + D_INNER + d];
    float v = (y_scan[idx] + Dvec[d] * xconv[idx]) * siluf(z);
    y_bf16[idx] = (bf16)v;
}

// ---------------------------------------------------------------------------
// orchestration
// ---------------------------------------------------------------------------
extern "C" void kernel_launch(void* const* d_in, const int* in_sizes, int n_in,
                              void* d_out, int out_size, void* d_ws, size_t ws_size,
                              hipStream_t stream) {
    const int*   tokens     = (const int*)  d_in[0];
    const float* emb        = (const float*)d_in[1];
    const float* in_proj_w  = (const float*)d_in[2];
    const float* conv_w     = (const float*)d_in[3];
    const float* conv_b     = (const float*)d_in[4];
    const float* x_proj_w   = (const float*)d_in[5];
    const float* dt_proj_w  = (const float*)d_in[6];
    const float* dt_proj_b  = (const float*)d_in[7];
    const float* A_log      = (const float*)d_in[8];
    const float* Dvec       = (const float*)d_in[9];
    const float* out_proj_w = (const float*)d_in[10];
    const float* head_w     = (const float*)d_in[11];
    const float* head_b     = (const float*)d_in[12];
    float* logits = (float*)d_out;

    char*  base = (char*)d_ws;
    size_t off  = 0;
    auto alloc = [&](size_t bytes) -> void* {
        off = (off + 255) & ~(size_t)255;
        void* p = base + off;
        off += bytes;
        return p;
    };

    bf16*  u_bf     = (bf16*) alloc((size_t)M_TOK * D_MODEL * 2);
    bf16*  inw_bf   = (bf16*) alloc((size_t)2 * D_INNER * D_MODEL * 2);
    float* xz       = (float*)alloc((size_t)M_TOK * 2 * D_INNER * 4);
    float* xconv    = (float*)alloc((size_t)M_TOK * D_INNER * 4);
    bf16*  x_bf     = (bf16*) alloc((size_t)M_TOK * D_INNER * 2);
    bf16*  xpw_bf   = (bf16*) alloc((size_t)96 * D_INNER * 2);
    float* xdbl     = (float*)alloc((size_t)M_TOK * 96 * 4);
    bf16*  xdbl_bf  = (bf16*) alloc((size_t)M_TOK * 96 * 2);
    bf16*  dtw_bf   = (bf16*) alloc((size_t)D_INNER * DT_RANK * 2);
    float* dt_arr   = (float*)alloc((size_t)M_TOK * D_INNER * 4);
    float* y_scan   = (float*)alloc((size_t)M_TOK * D_INNER * 4);
    bf16*  y_bf     = (bf16*) alloc((size_t)M_TOK * D_INNER * 2);
    bf16*  opw_bf   = (bf16*) alloc((size_t)D_MODEL * D_INNER * 2);
    float* out_f    = (float*)alloc((size_t)M_TOK * D_MODEL * 4);
    bf16*  out_bf   = (bf16*) alloc((size_t)M_TOK * D_MODEL * 2);
    bf16*  hw_bf    = (bf16*) alloc((size_t)VOCAB * D_MODEL * 2);
    (void)ws_size; (void)in_sizes; (void)n_in; (void)out_size;

    dim3 blk(256);
    auto nb = [](long long n) { return dim3((unsigned)((n + 255) / 256)); };

    // weight conversions f32 -> bf16
    cvt_bf16_kernel<<<nb((long long)2 * D_INNER * D_MODEL), blk, 0, stream>>>(in_proj_w, inw_bf, 2 * D_INNER * D_MODEL);
    cvt_bf16_kernel<<<nb((long long)96 * D_INNER),          blk, 0, stream>>>(x_proj_w,  xpw_bf, 96 * D_INNER);
    cvt_bf16_kernel<<<nb((long long)D_INNER * DT_RANK),     blk, 0, stream>>>(dt_proj_w, dtw_bf, D_INNER * DT_RANK);
    cvt_bf16_kernel<<<nb((long long)D_MODEL * D_INNER),     blk, 0, stream>>>(out_proj_w, opw_bf, D_MODEL * D_INNER);
    cvt_bf16_kernel<<<nb((long long)VOCAB * D_MODEL),       blk, 0, stream>>>(head_w,    hw_bf,  VOCAB * D_MODEL);

    // embedding gather
    embed_kernel<<<nb((long long)M_TOK * D_MODEL), blk, 0, stream>>>(tokens, emb, u_bf);

    // in_proj: [4096,1024] x [4096,1024]^T -> xz [4096,4096]
    wmma_gemm_kernel<0, true><<<dim3((2 * D_INNER) / BN, M_TOK / BM), blk, 0, stream>>>(
        u_bf, D_MODEL, inw_bf, D_MODEL, nullptr, xz, 2 * D_INNER, M_TOK, 2 * D_INNER, D_MODEL);

    // causal depthwise conv + SiLU
    conv_silu_kernel<<<nb((long long)M_TOK * D_INNER), blk, 0, stream>>>(xz, conv_w, conv_b, xconv, x_bf);

    // x_proj: [4096,2048] x [96,2048]^T -> xdbl [4096,96]  (guarded path)
    wmma_gemm_kernel<0, false><<<dim3(1, M_TOK / BM), blk, 0, stream>>>(
        x_bf, D_INNER, xpw_bf, D_INNER, nullptr, xdbl, 96, M_TOK, 96, D_INNER);

    // dt input (cols 0..63 of xdbl) consumed via bf16 copy of full xdbl (lda=96)
    cvt_bf16_kernel<<<nb((long long)M_TOK * 96), blk, 0, stream>>>(xdbl, xdbl_bf, M_TOK * 96);

    // dt_proj + bias + softplus: [4096,64] x [2048,64]^T -> dt [4096,2048]
    wmma_gemm_kernel<2, true><<<dim3(D_INNER / BN, M_TOK / BM), blk, 0, stream>>>(
        xdbl_bf, 96, dtw_bf, DT_RANK, dt_proj_b, dt_arr, D_INNER, M_TOK, D_INNER, DT_RANK);

    // selective scan
    scan_kernel<<<dim3((BATCH * D_INNER) / 16), blk, 0, stream>>>(dt_arr, xconv, xdbl, A_log, y_scan);

    // gate with silu(z), add D*x
    gate_kernel<<<nb((long long)M_TOK * D_INNER), blk, 0, stream>>>(y_scan, xconv, xz, Dvec, y_bf);

    // out_proj: [4096,2048] x [1024,2048]^T -> out [4096,1024]
    wmma_gemm_kernel<0, true><<<dim3(D_MODEL / BN, M_TOK / BM), blk, 0, stream>>>(
        y_bf, D_INNER, opw_bf, D_INNER, nullptr, out_f, D_MODEL, M_TOK, D_MODEL, D_INNER);

    cvt_bf16_kernel<<<nb((long long)M_TOK * D_MODEL), blk, 0, stream>>>(out_f, out_bf, M_TOK * D_MODEL);

    // head: [4096,1024] x [32000,1024]^T + bias -> logits [4096,32000]
    wmma_gemm_kernel<1, true><<<dim3(VOCAB / BN, M_TOK / BM), blk, 0, stream>>>(
        out_bf, D_MODEL, hw_bf, D_MODEL, head_b, logits, VOCAB, M_TOK, VOCAB, D_MODEL);
}
